// LAN_4320737100678
// MI455X (gfx1250) — matrix operand order repaired and checked by reading, not compiled, not run on hardware
//
#include <hip/hip_runtime.h>
#include <hip/hip_bf16.h>
#include <stdint.h>

// B=32, L=512, E=512, W=5, S=2.  All matmuls via v_wmma_f32_16x16x32_f16.
// - B operands pre-transposed to [N,K]: every fragment load is 32B contiguous.
// - GEMMs are software-pipelined (double-buffered fragments) with 2x4 register
//   blocking per wave (32x64 output tile, 8 WMMAs per 12 b128 loads).
// - GEMM1 uses a zero-padded m1 copy so window loads are branch-free.
typedef __attribute__((ext_vector_type(16))) _Float16 v16h;
typedef __attribute__((ext_vector_type(8)))  float    v8f;

#define BN_EPS 1e-3f
constexpr int Bb = 32, Ll = 512, Ee = 512, Ww = 5;
constexpr int Rr = Bb * Ll;      // 16384 rows (b,l)
constexpr int KF = Ww * Ee;      // 2560
constexpr int LP = Ll + 4;       // padded rows per batch (2 left + 2 right)

union AFrag { v16h h; uint32_t u[8]; };
struct Frags { AFrag a[2]; v16h b[4]; };

// ---------------- operand prep ----------------
__global__ __launch_bounds__(256) void pad_zero_f16(_Float16* __restrict__ m1p) {
  // zero rows {0,1,514,515} of each padded batch slab
  const int i = blockIdx.x * 256 + threadIdx.x;      // [0, 32*4*512)
  const int b = i >> 11, j = (i >> 9) & 3, e = i & 511;
  const int row = (j < 2) ? j : j + 512;             // 0,1,514,515
  m1p[((long long)(b * LP + row) << 9) + e] = (_Float16)0.f;
}

__global__ __launch_bounds__(256) void cvt_pad_f16(const float* __restrict__ m1,
                                                   _Float16* __restrict__ m1p) {
  const int i = blockIdx.x * 256 + threadIdx.x;      // [0, B*L*E)
  const int b = i >> 18, rem = i & 262143, l = rem >> 9, e = rem & 511;
  m1p[((long long)(b * LP + 2 + l) << 9) + e] = (_Float16)m1[i];
}

// f32 [R,C] -> f16 [C,R] transpose-convert (LDS tiled)
__global__ __launch_bounds__(256) void cvtT_f32_f16(const float* __restrict__ in,
                                                    _Float16* __restrict__ out,
                                                    int R, int C) {
  __shared__ _Float16 t[32][33];
  const int c0 = blockIdx.x * 32, r0 = blockIdx.y * 32;
  const int tx = threadIdx.x & 31, ty = threadIdx.x >> 5;
#pragma unroll
  for (int i = 0; i < 32; i += 8)
    t[ty + i][tx] = (_Float16)in[(long long)(r0 + ty + i) * C + c0 + tx];
  __syncthreads();
#pragma unroll
  for (int i = 0; i < 32; i += 8)
    out[(long long)(c0 + ty + i) * R + r0 + tx] = t[tx][ty + i];
}

// ---------------- WMMA fragment helpers ----------------
__device__ __forceinline__ void load_frags(const _Float16* a0, const _Float16* a1,
                                           const _Float16* const brow[4], int k0,
                                           Frags& F) {
  const uint32_t* p0 = (const uint32_t*)(a0 + k0);
  const uint32_t* p1 = (const uint32_t*)(a1 + k0);
#pragma unroll
  for (int i = 0; i < 4; ++i) {
    F.a[0].u[i] = p0[i]; F.a[0].u[i + 4] = p0[i + 4];
    F.a[1].u[i] = p1[i]; F.a[1].u[i + 4] = p1[i + 4];
  }
#pragma unroll
  for (int t = 0; t < 4; ++t) F.b[t] = *(const v16h*)(brow[t] + k0);
}

__device__ __forceinline__ void mma_frags(const Frags& F, v8f acc[2][4]) {
#pragma unroll
  for (int m = 0; m < 2; ++m)
#pragma unroll
    for (int t = 0; t < 4; ++t)
      acc[m][t] = __builtin_amdgcn_wmma_f32_16x16x32_f16(
          false, F.a[m].h, false, F.b[t], (short)0, acc[m][t], false, false);
}

// ---------------- WMMA GEMM, B stored [N,K] ----------------
// C[M,N] = A[M,K] * Bt[N,K]^T + bias[rowLocal, col].  Wave -> 32x64 tile.
__global__ __launch_bounds__(128)
void gemm_bt_wmma(const _Float16* __restrict__ A, const _Float16* __restrict__ Bt,
                  const float* __restrict__ bias, float* __restrict__ C,
                  int M, int N, int K, long long sA, long long sB, long long sC) {
  const int lane = threadIdx.x & 31;
  const int wv   = threadIdx.x >> 5;
  const int tm   = blockIdx.y * 4 + wv;      // 32-row tile index
  const int n0   = blockIdx.x * 64;
  const int z    = blockIdx.z;

  const _Float16* Az = A + (long long)z * sA;
  const _Float16* Bz = Bt + (long long)z * sB;
  float*          Cz = C + (long long)z * sC;

  const int mrow = lane & 15;
  const int hi   = lane >> 4;
  const int ksel = hi << 4;
  const int koff = hi << 4;
  const int ncol = lane & 15;

  const _Float16* a0 = Az + (long long)(tm * 32 + mrow) * K + ksel;
  const _Float16* a1 = a0 + (long long)16 * K;
  const _Float16* brow[4];
#pragma unroll
  for (int t = 0; t < 4; ++t)
    brow[t] = Bz + (long long)(n0 + t * 16 + ncol) * K + koff;

  v8f acc[2][4];
#pragma unroll
  for (int m = 0; m < 2; ++m)
#pragma unroll
    for (int t = 0; t < 4; ++t) acc[m][t] = (v8f){};

  Frags X0, X1;
  const int nSteps = K >> 5;                 // even for all our K
  load_frags(a0, a1, brow, 0, X0);
  int s = 1;
  for (; s < nSteps - 1; s += 2) {           // double-buffered pipeline
    load_frags(a0, a1, brow, s << 5, X1);
    mma_frags(X0, acc);
    load_frags(a0, a1, brow, (s + 1) << 5, X0);
    mma_frags(X1, acc);
  }
  load_frags(a0, a1, brow, s << 5, X1);
  mma_frags(X0, acc);
  mma_frags(X1, acc);

  const int rb = hi << 3;
#pragma unroll
  for (int m = 0; m < 2; ++m)
#pragma unroll
    for (int t = 0; t < 4; ++t) {
      const int col = n0 + t * 16 + ncol;
#pragma unroll
      for (int j = 0; j < 8; ++j) {
        const int mg = tm * 32 + m * 16 + rb + j;
        float v = acc[m][t][j];
        if (bias) v += bias[(mg & (Ll - 1)) * N + col];
        Cz[(long long)mg * N + col] = v;
      }
    }
}

// ---------------- GEMM1: branch-free sliding-window A, B = fT[E, 2560] ----------------
__global__ __launch_bounds__(128)
void gemm_win_wmma(const _Float16* __restrict__ m1p, const _Float16* __restrict__ fT,
                   const float* __restrict__ kb, float* __restrict__ C) {
  const int lane = threadIdx.x & 31;
  const int wv   = threadIdx.x >> 5;
  const int tm   = blockIdx.y * 4 + wv;
  const int n0   = blockIdx.x * 64;

  const int mrow = lane & 15;
  const int hi   = lane >> 4;
  const int ksel = hi << 4;
  const int koff = hi << 4;
  const int ncol = lane & 15;

  const int r0  = tm * 32 + mrow;            // 32-row tile never crosses a batch
  const int bz  = r0 >> 9;
  const int l0  = r0 & (Ll - 1);
  const long long row0 = (long long)bz * LP + l0;        // padded row for w=0
  const _Float16* A0 = m1p + (row0 << 9) + ksel;
  const _Float16* A1 = A0 + ((long long)16 << 9);

  const _Float16* brow[4];
#pragma unroll
  for (int t = 0; t < 4; ++t)
    brow[t] = fT + (long long)(n0 + t * 16 + ncol) * KF + koff;

  v8f acc[2][4];
#pragma unroll
  for (int m = 0; m < 2; ++m)
#pragma unroll
    for (int t = 0; t < 4; ++t) acc[m][t] = (v8f){};

  // A address for chunk k0: window slot w=k0>>9, e0=k0&511 -> row offset w, col e0
  auto achunk = [&](const _Float16* base, int k0) -> const _Float16* {
    const int w = k0 >> 9, e0 = k0 & (Ee - 1);
    return base + (((long long)w) << 9) + e0;
  };
  auto loadw = [&](int k0, Frags& F) {
    const uint32_t* p0 = (const uint32_t*)achunk(A0, k0);
    const uint32_t* p1 = (const uint32_t*)achunk(A1, k0);
#pragma unroll
    for (int i = 0; i < 4; ++i) {
      F.a[0].u[i] = p0[i]; F.a[0].u[i + 4] = p0[i + 4];
      F.a[1].u[i] = p1[i]; F.a[1].u[i + 4] = p1[i + 4];
    }
#pragma unroll
    for (int t = 0; t < 4; ++t) F.b[t] = *(const v16h*)(brow[t] + k0);
  };

  Frags X0, X1;
  const int nSteps = KF >> 5;                // 80
  loadw(0, X0);
  int s = 1;
  for (; s < nSteps - 1; s += 2) {
    loadw(s << 5, X1);
    mma_frags(X0, acc);
    loadw((s + 1) << 5, X0);
    mma_frags(X1, acc);
  }
  loadw(s << 5, X1);
  mma_frags(X0, acc);
  mma_frags(X1, acc);

  const int rb = hi << 3;
#pragma unroll
  for (int m = 0; m < 2; ++m)
#pragma unroll
    for (int t = 0; t < 4; ++t) {
      const int col = n0 + t * 16 + ncol;
#pragma unroll
      for (int j = 0; j < 8; ++j) {
        const int mg = tm * 32 + m * 16 + rb + j;
        C[(long long)mg * Ee + col] = acc[m][t][j] + kb[(mg & (Ll - 1)) * Ee + col];
      }
    }
}

// ---------------- deterministic per-channel BN stats ----------------
__global__ __launch_bounds__(256)
void col_partial(const float* __restrict__ y, float* __restrict__ ps,
                 float* __restrict__ pq, int C, int rowsPer) {
  const int c  = blockIdx.x * 256 + threadIdx.x;
  const int r0 = blockIdx.y * rowsPer;
  float s = 0.f, q = 0.f;
  for (int r = 0; r < rowsPer; ++r) {
    float v = y[(long long)(r0 + r) * C + c];
    s += v; q += v * v;
  }
  ps[blockIdx.y * C + c] = s;
  pq[blockIdx.y * C + c] = q;
}

__global__ __launch_bounds__(256)
void bn_finalize(const float* __restrict__ ps, const float* __restrict__ pq,
                 const float* __restrict__ g, const float* __restrict__ bt,
                 float* __restrict__ scl, float* __restrict__ shf,
                 int C, int nPart, float invR) {
  const int c = blockIdx.x * 256 + threadIdx.x;
  if (c >= C) return;
  float s = 0.f, q = 0.f;
  for (int p = 0; p < nPart; ++p) { s += ps[p * C + c]; q += pq[p * C + c]; }
  const float m  = s * invR;
  const float vv = q * invR - m * m;
  const float a  = g[c] * rsqrtf(vv + BN_EPS);
  scl[c] = a;
  shf[c] = bt[c] - m * a;
}

__global__ __launch_bounds__(256)
void bn_elu_to_h(const float* __restrict__ x, const float* __restrict__ scl,
                 const float* __restrict__ shf, _Float16* __restrict__ out, int n) {
  const int i = blockIdx.x * 256 + threadIdx.x;
  if (i >= n) return;
  const int c = i & (Ee - 1);
  float v = fmaf(scl[c], x[i], shf[c]);
  v = v > 0.f ? v : expm1f(v);
  out[i] = (_Float16)v;
}

__global__ __launch_bounds__(256)
void bn_elu_to_f(const float* __restrict__ x, const float* __restrict__ scl,
                 const float* __restrict__ shf, float* __restrict__ out, int n) {
  const int i = blockIdx.x * 256 + threadIdx.x;
  if (i >= n) return;
  const int c = i & (Ee - 1);
  float v = fmaf(scl[c], x[i], shf[c]);
  out[i] = v > 0.f ? v : expm1f(v);
}

// BN+ELU producing BOTH row-major lh[b,l,e] and per-batch transposed lhT[b,e,l].
__global__ __launch_bounds__(256)
void bn_elu_to_h_T(const float* __restrict__ y, const float* __restrict__ scl,
                   const float* __restrict__ shf, _Float16* __restrict__ lh,
                   _Float16* __restrict__ lhT) {
  __shared__ _Float16 t[32][33];
  const int b  = blockIdx.z;
  const int e0 = blockIdx.x * 32, l0 = blockIdx.y * 32;
  const int tx = threadIdx.x & 31, ty = threadIdx.x >> 5;
  const float* yb  = y   + ((long long)b << 18);
  _Float16*    lhb = lh  + ((long long)b << 18);
  _Float16*    lTb = lhT + ((long long)b << 18);
#pragma unroll
  for (int i = 0; i < 32; i += 8) {
    const int l = l0 + ty + i, e = e0 + tx;
    float v = fmaf(scl[e], yb[(long long)l * Ee + e], shf[e]);
    v = v > 0.f ? v : expm1f(v);
    const _Float16 h = (_Float16)v;
    lhb[(long long)l * Ee + e] = h;
    t[ty + i][tx] = h;
  }
  __syncthreads();
#pragma unroll
  for (int i = 0; i < 32; i += 8)
    lTb[(long long)(e0 + ty + i) * Ll + l0 + tx] = t[tx][ty + i];
}

// ---------------- softmax over axis=1 (queries) with fused BN affine ----------------
__global__ __launch_bounds__(256)
void softmax_q(const float* __restrict__ w, const float* __restrict__ scl,
               const float* __restrict__ shf, _Float16* __restrict__ wh) {
  const int idx = blockIdx.x * 256 + threadIdx.x;   // [0, B*L)
  const int b = idx >> 9;
  const int k = idx & (Ll - 1);
  const float* base = w + ((long long)b << 18) + k;
  const float a = scl[k], c = shf[k];
  float m = -3.4e38f;
  for (int q = 0; q < Ll; ++q) m = fmaxf(m, fmaf(a, base[(long long)q << 9], c));
  float s = 0.f;
  for (int q = 0; q < Ll; ++q) s += expf(fmaf(a, base[(long long)q << 9], c) - m);
  const float inv = 1.f / s;
  _Float16* dst = wh + ((long long)b << 18) + k;
  for (int q = 0; q < Ll; ++q) {
    float v = expf(fmaf(a, base[(long long)q << 9], c) - m) * inv;
    dst[(long long)q << 9] = (_Float16)v;
  }
}

// ---------------- launcher ----------------
extern "C" void kernel_launch(void* const* d_in, const int* in_sizes, int n_in,
                              void* d_out, int out_size, void* d_ws, size_t ws_size,
                              hipStream_t stream) {
  (void)in_sizes; (void)n_in; (void)out_size; (void)ws_size;
  const float* m1 = (const float*)d_in[0];
  const float* f  = (const float*)d_in[1];
  const float* wq = (const float*)d_in[2];
  const float* wk = (const float*)d_in[3];
  const float* qb = (const float*)d_in[4];
  const float* kb = (const float*)d_in[5];
  const float* wb = (const float*)d_in[6];
  const float* g1 = (const float*)d_in[7],  *b1 = (const float*)d_in[8];
  const float* g2 = (const float*)d_in[9],  *b2 = (const float*)d_in[10];
  const float* g3 = (const float*)d_in[11], *b3 = (const float*)d_in[12];
  const float* g4 = (const float*)d_in[13], *b4 = (const float*)d_in[14];
  const float* g5 = (const float*)d_in[15], *b5 = (const float*)d_in[16];
  float* out = (float*)d_out;

  char* ws = (char*)d_ws;
  size_t off = 0;
  auto take = [&](size_t bytes) { size_t r = off; off += (bytes + 255) & ~(size_t)255; return r; };
  _Float16* m1p  = (_Float16*)(ws + take((size_t)Bb * LP * Ee * 2)); // padded; later q2h
  _Float16* fT   = (_Float16*)(ws + take((size_t)KF * Ee * 2));      // [E, W*E]
  _Float16* wqT  = (_Float16*)(ws + take((size_t)Ee * Ee * 2));
  _Float16* wkT  = (_Float16*)(ws + take((size_t)Ee * Ee * 2));
  _Float16* lh   = (_Float16*)(ws + take((size_t)Rr * Ee * 2));      // [b,l,e]
  _Float16* lhT  = (_Float16*)(ws + take((size_t)Rr * Ee * 2));      // [b,e,l]
  float*    yA   = (float*)(ws + take((size_t)Rr * Ee * 4));         // y1, later w
  float*    yB   = (float*)(ws + take((size_t)Rr * Ee * 4));         // y2, later wh
  float*    yC   = (float*)(ws + take((size_t)Rr * Ee * 4));         // y3, later out
  _Float16* k2h  = (_Float16*)(ws + take((size_t)Rr * Ee * 2));
  float*    ps   = (float*)(ws + take((size_t)64 * Ee * 4));
  float*    pq   = (float*)(ws + take((size_t)64 * Ee * 4));
  float*    scl  = (float*)(ws + take((size_t)Ee * 4));
  float*    shf  = (float*)(ws + take((size_t)Ee * 4));
  _Float16* q2h  = m1p;                  // m1p dead after GEMM1
  _Float16* wh   = (_Float16*)yB;        // y2 dead after q2h is built

  const int NE = Rr * Ee;                // 8388608 elements
  const dim3 gEw((NE + 255) / 256), b256(256), b128(128);
  const long long sLE = (long long)Ll * Ee, sLL = (long long)Ll * Ll;

  auto stats = [&](const float* y, const float* g, const float* bt) {
    col_partial<<<dim3(2, 64), b256, 0, stream>>>(y, ps, pq, Ee, 256);
    bn_finalize<<<dim3(2), b256, 0, stream>>>(ps, pq, g, bt, scl, shf, Ee, 64, 1.0f / Rr);
  };

  // 0) operand prep: padded f16 m1; transposed f16 weights
  pad_zero_f16<<<dim3(Bb * 4 * Ee / 256), b256, 0, stream>>>(m1p);
  cvt_pad_f16<<<gEw, b256, 0, stream>>>(m1, m1p);
  cvtT_f32_f16<<<dim3(Ee / 32, KF / 32), b256, 0, stream>>>(f, fT, KF, Ee);
  cvtT_f32_f16<<<dim3(Ee / 32, Ee / 32), b256, 0, stream>>>(wq, wqT, Ee, Ee);
  cvtT_f32_f16<<<dim3(Ee / 32, Ee / 32), b256, 0, stream>>>(wk, wkT, Ee, Ee);

  // 1) y1 = unfold(m1) @ f + kb
  gemm_win_wmma<<<dim3(8, Rr / 128), b128, 0, stream>>>(m1p, fT, kb, yA);
  // 2) l = elu(BN1(y1)) -> lh + lhT
  stats(yA, g1, b1);
  bn_elu_to_h_T<<<dim3(Ee / 32, Ll / 32, Bb), b256, 0, stream>>>(yA, scl, shf, lh, lhT);

  // 3) y2 = l@wq + qb ; y3 = l@wk + kb
  gemm_bt_wmma<<<dim3(8, Rr / 128, 1), b128, 0, stream>>>(lh, wqT, qb, yB,
                                                          Rr, Ee, Ee, 0, 0, 0);
  gemm_bt_wmma<<<dim3(8, Rr / 128, 1), b128, 0, stream>>>(lh, wkT, kb, yC,
                                                          Rr, Ee, Ee, 0, 0, 0);
  // 4) q2, k2
  stats(yB, g2, b2);
  bn_elu_to_h<<<gEw, b256, 0, stream>>>(yB, scl, shf, q2h, NE);
  stats(yC, g3, b3);
  bn_elu_to_h<<<gEw, b256, 0, stream>>>(yC, scl, shf, k2h, NE);

  // 5) w = q2 @ k2^T + wb   (batched; k2h is naturally [N=k, K=e])
  gemm_bt_wmma<<<dim3(8, Ll / 128, Bb), b128, 0, stream>>>(q2h, k2h, wb, yA,
                                                           Ll, Ll, Ee, sLE, sLE, sLL);
  // 6) BN4 + softmax over q -> wh (f16)
  stats(yA, g4, b4);
  softmax_q<<<dim3(Rr / 256), b256, 0, stream>>>(yA, scl, shf, wh);

  // 7) out = w @ l   (batched; B operand = lhT[b,e,l])
  gemm_bt_wmma<<<dim3(8, Ll / 128, Bb), b128, 0, stream>>>(wh, lhT, nullptr, yC,
                                                           Ll, Ee, Ll, sLL, sLE, sLE);
  // 8) elu(BN5(out)) -> d_out (f32)
  stats(yC, g5, b5);
  bn_elu_to_f<<<gEw, b256, 0, stream>>>(yC, scl, shf, out, NE);
}
